// ExclusiveMultiHeadAttention_89601607729794
// MI455X (gfx1250) — matrix operand order repaired and checked by reading, not compiled, not run on hardware
//
#include <hip/hip_runtime.h>
#include <cstdint>

// ---------------------------------------------------------------------------
// ExclusiveMultiHeadAttention for MI455X (gfx1250), wave32 + WMMA bf16
// + async global->LDS staging (ASYNCcnt) in the flash-attention loop.
// B=2, S=2048, D=1024, H=16, HD=64.
// ---------------------------------------------------------------------------

typedef __bf16 bf16_t;
typedef __attribute__((ext_vector_type(16))) __bf16      v16bf;
typedef __attribute__((ext_vector_type(8)))  float        v8f;
typedef __attribute__((ext_vector_type(4)))  unsigned int v4u;
typedef __attribute__((ext_vector_type(4)))  int          v4i;

union BF16Frag { v16bf v; v4u q[2]; };
static_assert(sizeof(BF16Frag) == 32, "frag must be 8 VGPRs");

#define WMMA_BF16(a, b, c) \
  __builtin_amdgcn_wmma_f32_16x16x32_bf16(false, (a), false, (b), (short)0, (c), false, false)

constexpr int kB  = 2;
constexpr int kS  = 2048;
constexpr int kD  = 1024;
constexpr int kH  = 16;
constexpr int kHD = 64;
constexpr int kBS = kB * kS;       // 4096 rows
constexpr int kBH = kB * kH;       // 32 (b,h) pairs

// ---- CDNA5 async global->LDS copy (ASYNCcnt) ------------------------------
// Builtin parameter type (from hipcc diagnostic): pointer to int __vector(4).
typedef __attribute__((address_space(1))) v4i* gv4i_p;  // global src
typedef __attribute__((address_space(3))) v4i* lv4i_p;  // LDS dst

#if __has_builtin(__builtin_amdgcn_global_load_async_to_lds_b128)
#define ASYNC_B128(g, l) \
  __builtin_amdgcn_global_load_async_to_lds_b128((gv4i_p)(g), (lv4i_p)(l), 0, 0)
#else
#define ASYNC_B128(g, l)                                                   \
  asm volatile("global_load_async_to_lds_b128 %0, %1, off"                 \
               :: "v"((unsigned)(unsigned long)(lv4i_p)(l)),               \
                  "v"((unsigned long long)(size_t)(g))                     \
               : "memory")
#endif

__device__ __forceinline__ void wait_async0() {
#if __has_builtin(__builtin_amdgcn_s_wait_asynccnt)
  __builtin_amdgcn_s_wait_asynccnt(0);
#else
  asm volatile("s_wait_asynccnt 0x0" ::: "memory");
#endif
}

// ---- 16-lane XOR butterfly reductions via v_permlane16_b32 ----------------
#if __has_builtin(__builtin_amdgcn_permlane16)
__device__ __forceinline__ float permx16(float v, unsigned lo, unsigned hi) {
  return __uint_as_float(__builtin_amdgcn_permlane16(
      __float_as_uint(v), __float_as_uint(v), lo, hi, false, false));
}
__device__ __forceinline__ float redmax16(float v) {
  v = fmaxf(v, permx16(v, 0x67452301u, 0xEFCDAB89u));  // xor 1
  v = fmaxf(v, permx16(v, 0x54761032u, 0xDCFE98BAu));  // xor 2
  v = fmaxf(v, permx16(v, 0x32107654u, 0xBA98FEDCu));  // xor 4
  v = fmaxf(v, permx16(v, 0xFEDCBA98u, 0x76543210u));  // xor 8
  return v;
}
__device__ __forceinline__ float redsum16(float v) {
  v += permx16(v, 0x67452301u, 0xEFCDAB89u);
  v += permx16(v, 0x54761032u, 0xDCFE98BAu);
  v += permx16(v, 0x32107654u, 0xBA98FEDCu);
  v += permx16(v, 0xFEDCBA98u, 0x76543210u);
  return v;
}
#else
__device__ __forceinline__ float redmax16(float v) {
  for (int m = 1; m < 16; m <<= 1) v = fmaxf(v, __shfl_xor(v, m, 16));
  return v;
}
__device__ __forceinline__ float redsum16(float v) {
  for (int m = 1; m < 16; m <<= 1) v += __shfl_xor(v, m, 16);
  return v;
}
#endif

// ---------------------------------------------------------------------------
// 1) fp32 -> bf16 elementwise convert
// ---------------------------------------------------------------------------
__global__ void cvt_f32_bf16(const float* __restrict__ in, bf16_t* __restrict__ out, int n) {
  int i = blockIdx.x * blockDim.x + threadIdx.x;
  if (i < n) out[i] = (bf16_t)in[i];
}

// ---------------------------------------------------------------------------
// 2) W [1024 x 1024] fp32 (in-dim major) -> WT [n][k] bf16  (classic LDS tile)
// ---------------------------------------------------------------------------
__global__ __launch_bounds__(256) void transpose_cvt(const float* __restrict__ W,
                                                     bf16_t* __restrict__ WT) {
  __shared__ float tile[32][33];
  int n0 = blockIdx.x * 32, k0 = blockIdx.y * 32;
  for (int i = threadIdx.y; i < 32; i += 8)
    tile[i][threadIdx.x] = W[(size_t)(k0 + i) * kD + n0 + threadIdx.x];
  __syncthreads();
  for (int i = threadIdx.y; i < 32; i += 8)
    WT[(size_t)(n0 + i) * kD + k0 + threadIdx.x] = (bf16_t)tile[threadIdx.x][i];
}

// ---------------------------------------------------------------------------
// 3/6) bf16 WMMA GEMM: C[M=4096][N=1024] = A[m][k] * BT[n][k] + bias[n]
//   Block = 256 threads = 8 waves, tile 128(M) x 64(N); wave: 16x64 = 4 C-tiles.
//   MODE 0: store bf16 [bh][s][hd]        (Q, K)
//   MODE 1: MODE0 + bf16 vT [bh][hd][s]   (V)
//   MODE 2: store fp32 [m][n]             (final output projection)
// ---------------------------------------------------------------------------
template <int MODE>
__global__ __launch_bounds__(256) void gemm_bf16(const bf16_t* __restrict__ A,
                                                 const bf16_t* __restrict__ BT,
                                                 const float* __restrict__ bias,
                                                 void* __restrict__ out0,
                                                 void* __restrict__ out1) {
  const int wave = threadIdx.x >> 5;
  const int lane = threadIdx.x & 31;
  const int half = lane >> 4;
  const int l16  = lane & 15;

  const int mtile = blockIdx.y * 128 + wave * 16;  // 16 rows per wave
  const int nb    = blockIdx.x * 64;               // 4 n-tiles per wave

  v8f acc[4];
#pragma unroll
  for (int t = 0; t < 4; ++t)
#pragma unroll
    for (int r = 0; r < 8; ++r) acc[t][r] = 0.0f;

  const bf16_t* arow = A + (size_t)(mtile + l16) * kD;

  for (int kk = 0; kk < kD; kk += 32) {
    BF16Frag af;  // A 16x32: lane row = l16; runs at k = kk+half*8 and +16
    af.q[0] = *(const v4u*)(arow + kk + half * 8);
    af.q[1] = *(const v4u*)(arow + kk + half * 8 + 16);
#pragma unroll
    for (int t = 0; t < 4; ++t) {
      BF16Frag bfr;  // B 32x16: lane col = nb+t*16+l16; runs at k = kk+half*8 and +16
      const bf16_t* bp = BT + (size_t)(nb + t * 16 + l16) * kD + kk + half * 8;
      bfr.q[0] = *(const v4u*)(bp);
      bfr.q[1] = *(const v4u*)(bp + 16);
      acc[t] = WMMA_BF16(af.v, bfr.v, acc[t]);
    }
  }

#pragma unroll
  for (int t = 0; t < 4; ++t) {
    const int   n  = nb + t * 16 + l16;
    const float bv = bias[n];
#pragma unroll
    for (int r = 0; r < 8; ++r) {
      const int   m   = mtile + r + half * 8;
      const float val = acc[t][r] + bv;
      if (MODE == 2) {
        ((float*)out0)[(size_t)m * kD + n] = val;
      } else {
        const int    b  = m >> 11, s = m & (kS - 1);
        const int    h  = n >> 6, hd = n & (kHD - 1);
        const size_t bh = (size_t)b * kH + h;
        ((bf16_t*)out0)[(bh * kS + s) * kHD + hd] = (bf16_t)val;
        if (MODE == 1) ((bf16_t*)out1)[(bh * kHD + hd) * kS + s] = (bf16_t)val;
      }
    }
  }
}

// ---------------------------------------------------------------------------
// 4) Flash attention. Grid (S/64, B*H); block = 128 threads = 4 waves.
//    K/V tiles (32 keys) staged to LDS with double-buffered async copies;
//    each wave: 16 query rows, online softmax; 8 WMMAs per key block.
//    Q,K: bf16 [bh][s][hd];  vT: bf16 [bh][hd][s];  Y: fp32 [bh][s][hd].
// ---------------------------------------------------------------------------
__global__ __launch_bounds__(128) void flash_attn(const bf16_t* __restrict__ Q,
                                                  const bf16_t* __restrict__ Km,
                                                  const bf16_t* __restrict__ vT,
                                                  float* __restrict__ Y) {
  __shared__ __align__(16) bf16_t Kt[2][32][64];    // [buf][key][hd]    4KB each
  __shared__ __align__(16) bf16_t Vt[2][64][32];    // [buf][hd][key]    4KB each
  __shared__ __align__(16) bf16_t Plds[4][16][32];  // per-wave P round-trip

  const int tid  = threadIdx.x;
  const int wave = tid >> 5;
  const int lane = tid & 31;
  const int half = lane >> 4;
  const int l16  = lane & 15;

  const int    bh    = blockIdx.y;
  const int    qtile = blockIdx.x * 64 + wave * 16;
  const size_t base  = (size_t)bh * kS;

  // async-stage one 32-key block (K: 32x64, V: 64x32) -> LDS buffer `buf`
  auto stage = [&](int buf, int kb) {
#pragma unroll
    for (int c2 = 0; c2 < 2; ++c2) {
      const int c = tid + c2 * 128;  // 256 x 16B chunks for each of K and V
      const int krow = c >> 3, koff = (c & 7) * 8;
      ASYNC_B128(Km + (base + kb + krow) * kHD + koff, &Kt[buf][krow][koff]);
      const int vrow = c >> 2, voff = (c & 3) * 8;
      ASYNC_B128(vT + ((size_t)bh * kHD + vrow) * kS + kb + voff, &Vt[buf][vrow][voff]);
    }
  };

  // Q A-fragments covering hd 0..31 and 32..63 (lane row = l16)
  BF16Frag qf0, qf1;
  {
    const bf16_t* qp = Q + (base + qtile + l16) * kHD;
    qf0.q[0] = *(const v4u*)(qp + half * 8);
    qf0.q[1] = *(const v4u*)(qp + half * 8 + 16);
    qf1.q[0] = *(const v4u*)(qp + 32 + half * 8);
    qf1.q[1] = *(const v4u*)(qp + 32 + half * 8 + 16);
  }

  float m_i[8], l_i[8];
  v8f   o[4];
#pragma unroll
  for (int r = 0; r < 8; ++r) { m_i[r] = -1e30f; l_i[r] = 0.0f; }
#pragma unroll
  for (int t = 0; t < 4; ++t)
#pragma unroll
    for (int r = 0; r < 8; ++r) o[t][r] = 0.0f;

  const float scale = 0.125f;  // 1/sqrt(64)

  int buf = 0;
  stage(0, 0);

  for (int kb = 0; kb < kS; kb += 32) {
    wait_async0();     // own async copies done
    __syncthreads();   // everyone's copies visible
    if (kb + 32 < kS) stage(buf ^ 1, kb + 32);  // overlap next DMA with compute

    // ---- scores S = (q . k^T) * scale : two 16x16 tiles from LDS K ----
    v8f s[2];
#pragma unroll
    for (int j = 0; j < 2; ++j) {
      const bf16_t* kp = &Kt[buf][j * 16 + l16][0];  // lane = key col
      BF16Frag kb0, kb1;
      kb0.q[0] = *(const v4u*)(kp + half * 8);       // hd 0..31: runs at half*8 and +16
      kb0.q[1] = *(const v4u*)(kp + half * 8 + 16);
      kb1.q[0] = *(const v4u*)(kp + 32 + half * 8);  // hd 32..63: runs at 32+half*8 and +16
      kb1.q[1] = *(const v4u*)(kp + 32 + half * 8 + 16);
      v8f sa;
#pragma unroll
      for (int r = 0; r < 8; ++r) sa[r] = 0.0f;
      sa   = WMMA_BF16(qf0.v, kb0.v, sa);
      sa   = WMMA_BF16(qf1.v, kb1.v, sa);
      s[j] = sa;
    }

    // ---- online softmax update (rows r+half*8; cols across 16 lanes) ----
#pragma unroll
    for (int r = 0; r < 8; ++r) {
      float s0 = s[0][r] * scale, s1 = s[1][r] * scale;
      const float mx    = redmax16(fmaxf(s0, s1));
      const float mnew  = fmaxf(m_i[r], mx);
      const float alpha = __expf(m_i[r] - mnew);
      m_i[r] = mnew;
      const float p0 = __expf(s0 - mnew);
      const float p1 = __expf(s1 - mnew);
      s[0][r] = p0;
      s[1][r] = p1;
      l_i[r] = l_i[r] * alpha + redsum16(p0 + p1);
#pragma unroll
      for (int t = 0; t < 4; ++t) o[t][r] *= alpha;
    }

    // ---- P: C-layout -> A-layout via per-wave LDS round trip ----
#pragma unroll
    for (int j = 0; j < 2; ++j)
#pragma unroll
      for (int r = 0; r < 8; ++r)
        Plds[wave][r + half * 8][j * 16 + l16] = (bf16_t)s[j][r];

    asm volatile("s_wait_dscnt 0" ::: "memory");  // wave-local LDS RAW fence

    BF16Frag pf;  // A 16x32 over 32 keys
    {
      const bf16_t* pl = &Plds[wave][l16][0];
      pf.q[0] = *(const v4u*)(pl + half * 8);
      pf.q[1] = *(const v4u*)(pl + half * 8 + 16);
    }

    // ---- O += P . V : 4 hd n-tiles, V from LDS Vt[hd][key] ----
#pragma unroll
    for (int t = 0; t < 4; ++t) {
      const bf16_t* vp = &Vt[buf][t * 16 + l16][half * 8];
      BF16Frag vb;
      vb.q[0] = *(const v4u*)(vp);
      vb.q[1] = *(const v4u*)(vp + 16);
      o[t] = WMMA_BF16(pf.v, vb.v, o[t]);
    }

    __syncthreads();  // all waves done reading buf before it is re-staged
    buf ^= 1;
  }

  // ---- normalize and store Y fp32 [bh][s][hd] ----
#pragma unroll
  for (int r = 0; r < 8; ++r) {
    const float inv  = 1.0f / l_i[r];
    const int   qrow = qtile + r + half * 8;
#pragma unroll
    for (int t = 0; t < 4; ++t)
      Y[(base + qrow) * kHD + t * 16 + l16] = o[t][r] * inv;
  }
}

// ---------------------------------------------------------------------------
// 5) y -= (y . v_hat) v_hat, v_hat = v / ||v||; write bf16 [m=bs][n=h*64+hd].
//    One wave per (bh,s) row of 64; 2 elements per lane.
// ---------------------------------------------------------------------------
__global__ __launch_bounds__(128) void vproj_sub(const float* __restrict__ Y,
                                                 const bf16_t* __restrict__ V,
                                                 bf16_t* __restrict__ Yp) {
  const int wave = threadIdx.x >> 5;
  const int lane = threadIdx.x & 31;
  const int row  = blockIdx.x * 4 + wave;  // row = bh*S + s, 65536 rows

  const size_t off = (size_t)row * kHD;
  const float  y0  = Y[off + lane];
  const float  y1  = Y[off + lane + 32];
  const float  v0  = (float)V[off + lane];
  const float  v1  = (float)V[off + lane + 32];

  float vv = v0 * v0 + v1 * v1;
  float yv = y0 * v0 + y1 * v1;
#pragma unroll
  for (int m = 1; m < 32; m <<= 1) {
    vv += __shfl_xor(vv, m, 32);
    yv += __shfl_xor(yv, m, 32);
  }
  const float coef = yv / (vv + 1e-12f);

  const int    bh = row >> 11, s = row & (kS - 1);
  const int    b  = bh >> 4, h = bh & (kH - 1);
  const size_t m0 = (size_t)(b * kS + s) * kD + h * kHD;
  Yp[m0 + lane]      = (bf16_t)(y0 - coef * v0);
  Yp[m0 + lane + 32] = (bf16_t)(y1 - coef * v1);
}

// ---------------------------------------------------------------------------
// launch
// ---------------------------------------------------------------------------
extern "C" void kernel_launch(void* const* d_in, const int* in_sizes, int n_in,
                              void* d_out, int out_size, void* d_ws, size_t ws_size,
                              hipStream_t stream) {
  const float* x  = (const float*)d_in[0];
  const float* Wq = (const float*)d_in[1];
  const float* bq = (const float*)d_in[2];
  const float* Wk = (const float*)d_in[3];
  const float* bk = (const float*)d_in[4];
  const float* Wv = (const float*)d_in[5];
  const float* bv = (const float*)d_in[6];
  const float* Wo = (const float*)d_in[7];
  const float* bo = (const float*)d_in[8];
  float*       out = (float*)d_out;

  // workspace layout (bytes, all 256-aligned); total ~72 MB
  char*   ws  = (char*)d_ws;
  bf16_t* xb  = (bf16_t*)(ws);                         // 8 MB  [4096][1024]
  bf16_t* WqT = (bf16_t*)(ws + (8u << 20));            // 2 MB  [n][k]
  bf16_t* WkT = (bf16_t*)(ws + (10u << 20));           // 2 MB
  bf16_t* WvT = (bf16_t*)(ws + (12u << 20));           // 2 MB
  bf16_t* WoT = (bf16_t*)(ws + (14u << 20));           // 2 MB
  bf16_t* Qb  = (bf16_t*)(ws + (16u << 20));           // 8 MB  [bh][s][hd]
  bf16_t* Kb  = (bf16_t*)(ws + (24u << 20));           // 8 MB  [bh][s][hd]
  bf16_t* Vb  = (bf16_t*)(ws + (32u << 20));           // 8 MB  [bh][s][hd]
  bf16_t* VbT = (bf16_t*)(ws + (40u << 20));           // 8 MB  [bh][hd][s]
  float*  Yf  = (float*)(ws + (48u << 20));            // 16 MB [bh][s][hd]
  bf16_t* Yp  = (bf16_t*)(ws + (64u << 20));           // 8 MB  [bs][hk]

  const dim3 gemmGrid(kD / 64, kBS / 128), gemmBlk(256);
  const dim3 trGrid(kD / 32, kD / 32), trBlk(32, 8);

  // 1) x -> bf16
  cvt_f32_bf16<<<(kBS * kD + 255) / 256, 256, 0, stream>>>(x, xb, kBS * kD);

  // 2) weight transposes (+ bf16 convert)
  transpose_cvt<<<trGrid, trBlk, 0, stream>>>(Wq, WqT);
  transpose_cvt<<<trGrid, trBlk, 0, stream>>>(Wk, WkT);
  transpose_cvt<<<trGrid, trBlk, 0, stream>>>(Wv, WvT);
  transpose_cvt<<<trGrid, trBlk, 0, stream>>>(Wo, WoT);

  // 3) QKV projections (WMMA bf16)
  gemm_bf16<0><<<gemmGrid, gemmBlk, 0, stream>>>(xb, WqT, bq, Qb, nullptr);
  gemm_bf16<0><<<gemmGrid, gemmBlk, 0, stream>>>(xb, WkT, bk, Kb, nullptr);
  gemm_bf16<1><<<gemmGrid, gemmBlk, 0, stream>>>(xb, WvT, bv, Vb, VbT);

  // 4) flash attention (async-LDS double-buffered K/V)
  flash_attn<<<dim3(kS / 64, kBH), 128, 0, stream>>>(Qb, Kb, VbT, Yf);

  // 5) subtract projection onto normalized v
  vproj_sub<<<(kBH * kS) / 4, 128, 0, stream>>>(Yf, Vb, Yp);

  // 6) output projection (WMMA bf16, fp32 out + bias)
  gemm_bf16<2><<<gemmGrid, gemmBlk, 0, stream>>>(Yp, WoT, bo, out, nullptr);
}
